// NNUENetwork_34419867910599
// MI455X (gfx1250) — matrix operand order repaired and checked by reading, not compiled, not run on hardware
//
#include <hip/hip_runtime.h>

typedef __attribute__((ext_vector_type(16))) _Float16 v16h;
typedef __attribute__((ext_vector_type(8)))  _Float16 v8h;
typedef __attribute__((ext_vector_type(8)))  float    v8f;
typedef __attribute__((ext_vector_type(4)))  float    v4f;

#define FEATS 40960
#define H1 512
#define H2 256
#define BATCH 4096

#define BM 128
#define BN 128
#define KC 64

// ---------------------------------------------------------------------------
// Kernel 1: feature transform. white+black fused as one GEMM with M = 8192.
// C[m, n] = clamp( sum_k A[m,k] * ft_w[n,k] + ft_b[n], 0, 1 ), stored f16 into
// combined[4096][1024]  (white -> cols 0..511, black -> cols 512..1023).
// Per-wave 32x64 tile (2x4 fragments): 12 ds_load_b128 per 8 WMMAs.
// ---------------------------------------------------------------------------
__global__ __launch_bounds__(256) void ft_gemm_kernel(
    const float* __restrict__ wfeat, const float* __restrict__ bfeat,
    const float* __restrict__ ftw,   const float* __restrict__ ftb,
    _Float16* __restrict__ comb)
{
    __shared__ alignas(16) _Float16 smem[2 * BM * KC + 2 * BN * KC]; // 64 KB
    _Float16* As = smem;                  // [2][BM*KC]
    _Float16* Bs = smem + 2 * BM * KC;    // [2][BN*KC]

    const int tid  = threadIdx.x;
    const int lane = tid & 31;
    const int wave = tid >> 5;

    const int  mTile = blockIdx.x >> 2;      // 64 M-tiles
    const int  nTile = blockIdx.x & 3;       // 4  N-tiles (adjacent -> L2 reuse of A)
    const long m0    = (long)mTile * BM;     // 0..8191
    const int  n0    = nTile * BN;

    const float* Abase = (m0 < BATCH) ? wfeat : bfeat;
    const long   aRow0 = (m0 < BATCH) ? m0 : (m0 - BATCH);

    // Cooperative tile load mapping: 256 threads, each owns half a row (32 f32).
    const int ldRow = tid >> 1;
    const int ldCol = (tid & 1) * 32;

    const float* aPtr = Abase + (aRow0 + ldRow) * (long)FEATS + ldCol;
    const float* bPtr = ftw   + (long)(n0 + ldRow) * FEATS + ldCol;

    // Wave tile: waves arranged 4 (M) x 2 (N) over the 128x128 block tile.
    const int mBase = (wave & 3) * 32;
    const int nBase = (wave >> 2) * 64;

    v4f ar[8], br[8];
    v8f acc[2][4];
    const v8f vzero = {};
#pragma unroll
    for (int mi = 0; mi < 2; ++mi)
#pragma unroll
        for (int n = 0; n < 4; ++n) acc[mi][n] = vzero;

    const int NIT = FEATS / KC; // 640

    auto loadG = [&](int kt) {
        const v4f* a = (const v4f*)(aPtr + kt * KC);
        const v4f* b = (const v4f*)(bPtr + kt * KC);
#pragma unroll
        for (int i = 0; i < 8; ++i) { ar[i] = a[i]; br[i] = b[i]; }
    };

    auto storeL = [&](int buf) {
        v8h* ad = (v8h*)&As[buf * BM * KC + ldRow * KC + ldCol];
        v8h* bd = (v8h*)&Bs[buf * BN * KC + ldRow * KC + ldCol];
#pragma unroll
        for (int i = 0; i < 4; ++i) {
            v8h pa, pb;
#pragma unroll
            for (int j = 0; j < 4; ++j) {
                pa[j]     = (_Float16)ar[2 * i][j];
                pa[4 + j] = (_Float16)ar[2 * i + 1][j];
                pb[j]     = (_Float16)br[2 * i][j];
                pb[4 + j] = (_Float16)br[2 * i + 1][j];
            }
            ad[i] = pa; bd[i] = pb;
        }
    };

    auto compute = [&](int buf) {
        const _Float16* A = &As[buf * BM * KC];
        const _Float16* B = &Bs[buf * BN * KC];
#pragma unroll
        for (int s = 0; s < 2; ++s) {
            // 16-bit A 16x32 layout: lanes<16 hold K 0-7 & 16-23, lanes>=16 hold
            // K 8-15 & 24-31 (ISA 7.12.2).
            const int ka = (lane >> 4) * 8 + s * 32;
            v16h af[2];
#pragma unroll
            for (int mi = 0; mi < 2; ++mi) {
                const int mrow = mBase + mi * 16 + (lane & 15);
                v8h alo = *(const v8h*)&A[mrow * KC + ka];
                v8h ahi = *(const v8h*)&A[mrow * KC + ka + 16];
                af[mi] = __builtin_shufflevector(alo, ahi,
                    0, 1, 2, 3, 4, 5, 6, 7, 8, 9, 10, 11, 12, 13, 14, 15);
            }
            // 16-bit B 32x16 layout: lanes<16 hold K 0-7 & 16-23, lanes>=16 K 8-15 & 24-31.
            const int kb = (lane >> 4) * 8 + s * 32;
            v16h bfr[4];
#pragma unroll
            for (int n = 0; n < 4; ++n) {
                const int ncol = nBase + n * 16 + (lane & 15);
                v8h blo = *(const v8h*)&B[ncol * KC + kb];
                v8h bhi = *(const v8h*)&B[ncol * KC + kb + 16];
                bfr[n] = __builtin_shufflevector(blo, bhi,
                    0, 1, 2, 3, 4, 5, 6, 7, 8, 9, 10, 11, 12, 13, 14, 15);
            }
#pragma unroll
            for (int mi = 0; mi < 2; ++mi)
#pragma unroll
                for (int n = 0; n < 4; ++n)
                    acc[mi][n] = __builtin_amdgcn_wmma_f32_16x16x32_f16(
                        false, af[mi], false, bfr[n], (short)0, acc[mi][n],
                        false, false);
        }
    };

    loadG(0);
    storeL(0);
    for (int kt = 0; kt < NIT; ++kt) {
        __syncthreads();                     // buf[kt&1] visible; prev reads done
        if (kt + 1 < NIT) loadG(kt + 1);     // prefetch next chunk (overlaps WMMA)
        compute(kt & 1);
        if (kt + 1 < NIT) storeL((kt + 1) & 1);
    }
    __syncthreads();

    // Epilogue: bias + clamp to [0,1], stage f16 tile in LDS for coalesced store.
    _Float16* stg = smem;                    // 128x128 f16 = 32 KB (aliases As)
#pragma unroll
    for (int mi = 0; mi < 2; ++mi) {
        const int rowBase = mBase + mi * 16 + (lane >> 4) * 8;
#pragma unroll
        for (int n = 0; n < 4; ++n) {
            const int col  = nBase + n * 16 + (lane & 15);
            const float bv = ftb[n0 + col];
#pragma unroll
            for (int r = 0; r < 8; ++r) {
                float v = acc[mi][n][r] + bv;
                v = fminf(fmaxf(v, 0.0f), 1.0f);
                stg[(rowBase + r) * BN + col] = (_Float16)v;
            }
        }
    }
    __syncthreads();
    {
        const int  row = tid >> 1;
        const int  c0  = (tid & 1) * 64;     // 64 halves = 128 B per thread
        const long gm  = m0 + row;
        _Float16* dst = (gm < BATCH)
            ? comb + gm * (long)(2 * H1) + n0 + c0
            : comb + (gm - BATCH) * (long)(2 * H1) + H1 + n0 + c0;
        const v8h* s = (const v8h*)&stg[row * BN + c0];
        v8h*       d = (v8h*)dst;
#pragma unroll
        for (int i = 0; i < 8; ++i) d[i] = s[i];
    }
}

// ---------------------------------------------------------------------------
// Kernel 2: FC1.  x[m,n] = relu( sum_k combined[m,k]*fc1_w[n,k] + fc1_b[n] )
// combined is f16 [4096][1024] -> copied global->LDS with async-to-LDS DMA
// (pure f16 copy, no conversion needed; tracked on ASYNCcnt, zero VGPR
// staging). fc1_w f32 [256][1024] converted in registers. Output f32.
// ---------------------------------------------------------------------------
__global__ __launch_bounds__(256) void fc1_gemm_kernel(
    const _Float16* __restrict__ comb, const float* __restrict__ w,
    const float* __restrict__ bias,    float* __restrict__ xout)
{
    __shared__ alignas(16) char sraw[65536];
    _Float16* As = (_Float16*)sraw;              // [2][128*64] f16 = 32 KB @ LDS 0
    _Float16* Bs = (_Float16*)(sraw + 32768);    // [2][128*64] f16 = 32 KB

    const int tid  = threadIdx.x;
    const int lane = tid & 31;
    const int wave = tid >> 5;

    const int mTile = blockIdx.x >> 1;           // 32 M-tiles
    const int nTile = blockIdx.x & 1;            // 2  N-tiles
    const int m0 = mTile * BM;
    const int n0 = nTile * BN;

    const int ldRow = tid >> 1;
    const int ldCol = (tid & 1) * 32;

    const _Float16* aPtr = comb + (long)(m0 + ldRow) * (2 * H1) + ldCol;
    const float*    bPtr = w    + (long)(n0 + ldRow) * (2 * H1) + ldCol;

    const int mBase = (wave & 3) * 32;
    const int nBase = (wave >> 2) * 64;

    v4f br[8];
    v8f acc[2][4];
    const v8f vzero = {};
#pragma unroll
    for (int mi = 0; mi < 2; ++mi)
#pragma unroll
        for (int n = 0; n < 4; ++n) acc[mi][n] = vzero;

    const int NIT = (2 * H1) / KC;               // 16

    // Async DMA: global f16 tile -> LDS, 4 x b128 per thread, ASYNCcnt-tracked.
    auto issueAsyncA = [&](int kt, int buf) {
        const _Float16* g = aPtr + kt * KC;
        const unsigned ldsByte =
            (unsigned)((buf * BM * KC + ldRow * KC + ldCol) * 2); // As @ LDS 0
#pragma unroll
        for (int i = 0; i < 4; ++i) {
            unsigned dst = ldsByte + i * 16;
            unsigned long long src = (unsigned long long)g + (unsigned)(i * 16);
            asm volatile("global_load_async_to_lds_b128 %0, %1, off"
                         :: "v"(dst), "v"(src) : "memory");
        }
    };

    auto loadGB = [&](int kt) {
        const v4f* b = (const v4f*)(bPtr + kt * KC);
#pragma unroll
        for (int i = 0; i < 8; ++i) br[i] = b[i];
    };

    auto storeLB = [&](int buf) {
        v8h* bd = (v8h*)&Bs[buf * BM * KC + ldRow * KC + ldCol];
#pragma unroll
        for (int i = 0; i < 4; ++i) {
            v8h pb;
#pragma unroll
            for (int j = 0; j < 4; ++j) {
                pb[j]     = (_Float16)br[2 * i][j];
                pb[4 + j] = (_Float16)br[2 * i + 1][j];
            }
            bd[i] = pb;
        }
    };

    auto compute = [&](int buf) {
        const _Float16* A = &As[buf * BM * KC];
        const _Float16* B = &Bs[buf * BM * KC];
#pragma unroll
        for (int s = 0; s < 2; ++s) {
            const int ka = (lane >> 4) * 8 + s * 32;
            v16h af[2];
#pragma unroll
            for (int mi = 0; mi < 2; ++mi) {
                const int mrow = mBase + mi * 16 + (lane & 15);
                v8h alo = *(const v8h*)&A[mrow * KC + ka];
                v8h ahi = *(const v8h*)&A[mrow * KC + ka + 16];
                af[mi] = __builtin_shufflevector(alo, ahi,
                    0, 1, 2, 3, 4, 5, 6, 7, 8, 9, 10, 11, 12, 13, 14, 15);
            }
            const int kb = (lane >> 4) * 8 + s * 32;
            v16h bfr[4];
#pragma unroll
            for (int n = 0; n < 4; ++n) {
                const int ncol = nBase + n * 16 + (lane & 15);
                v8h blo = *(const v8h*)&B[ncol * KC + kb];
                v8h bhi = *(const v8h*)&B[ncol * KC + kb + 16];
                bfr[n] = __builtin_shufflevector(blo, bhi,
                    0, 1, 2, 3, 4, 5, 6, 7, 8, 9, 10, 11, 12, 13, 14, 15);
            }
#pragma unroll
            for (int mi = 0; mi < 2; ++mi)
#pragma unroll
                for (int n = 0; n < 4; ++n)
                    acc[mi][n] = __builtin_amdgcn_wmma_f32_16x16x32_f16(
                        false, af[mi], false, bfr[n], (short)0, acc[mi][n],
                        false, false);
        }
    };

    issueAsyncA(0, 0);
    loadGB(0);
    storeLB(0);
    for (int kt = 0; kt < NIT; ++kt) {
        asm volatile("s_wait_asynccnt 0" ::: "memory"); // A tile landed in LDS
        __syncthreads();
        if (kt + 1 < NIT) {
            issueAsyncA(kt + 1, (kt + 1) & 1);          // DMA next A in background
            loadGB(kt + 1);
        }
        compute(kt & 1);
        if (kt + 1 < NIT) storeLB((kt + 1) & 1);
    }
    __syncthreads();

    // relu(acc + bias), stage f32 tile (64 KB, aliases whole smem), coalesced out
    float* stg = (float*)sraw;
#pragma unroll
    for (int mi = 0; mi < 2; ++mi) {
        const int rowBase = mBase + mi * 16 + (lane >> 4) * 8;
#pragma unroll
        for (int n = 0; n < 4; ++n) {
            const int  col = nBase + n * 16 + (lane & 15);
            const float bv = bias[n0 + col];
#pragma unroll
            for (int r = 0; r < 8; ++r) {
                float v = acc[mi][n][r] + bv;
                stg[(rowBase + r) * BN + col] = fmaxf(v, 0.0f);
            }
        }
    }
    __syncthreads();
    {
        const int row = tid >> 1;
        const int c0  = (tid & 1) * 64;          // 64 f32 = 256 B per thread
        float* dst = xout + (long)(m0 + row) * H2 + n0 + c0;
        const v4f* s = (const v4f*)&stg[row * BN + c0];
        v4f*       d = (v4f*)dst;
#pragma unroll
        for (int i = 0; i < 16; ++i) d[i] = s[i];
    }
}

// ---------------------------------------------------------------------------
// Kernel 3: FC2.  out[m] = dot(x[m,:], fc2_w) + fc2_b.  One wave per row,
// shuffle reduction (deterministic, no FP atomics).
// ---------------------------------------------------------------------------
__global__ __launch_bounds__(256) void fc2_dot_kernel(
    const float* __restrict__ x, const float* __restrict__ w,
    const float* __restrict__ b, float* __restrict__ out)
{
    const int lane = threadIdx.x & 31;
    const int wave = threadIdx.x >> 5;
    const int m    = blockIdx.x * 8 + wave;

    const v4f* xr = (const v4f*)(x + (long)m * H2 + lane * 8);
    const v4f* wr = (const v4f*)(w + lane * 8);
    v4f x0 = xr[0], x1 = xr[1];
    v4f w0 = wr[0], w1 = wr[1];
    float s = 0.0f;
#pragma unroll
    for (int i = 0; i < 4; ++i) s += x0[i] * w0[i] + x1[i] * w1[i];

#pragma unroll
    for (int off = 16; off > 0; off >>= 1) s += __shfl_down(s, off, 32);

    if (lane == 0) out[m] = s + b[0];
}

// ---------------------------------------------------------------------------
extern "C" void kernel_launch(void* const* d_in, const int* in_sizes, int n_in,
                              void* d_out, int out_size, void* d_ws, size_t ws_size,
                              hipStream_t stream)
{
    const float* wfeat = (const float*)d_in[0];
    const float* bfeat = (const float*)d_in[1];
    const float* ftw   = (const float*)d_in[2];
    const float* ftb   = (const float*)d_in[3];
    const float* fc1w  = (const float*)d_in[4];
    const float* fc1b  = (const float*)d_in[5];
    const float* fc2w  = (const float*)d_in[6];
    const float* fc2b  = (const float*)d_in[7];

    _Float16* comb = (_Float16*)d_ws;                               // 8 MB f16
    float*    xbuf = (float*)((char*)d_ws +
                       (size_t)BATCH * (2 * H1) * sizeof(_Float16)); // 4 MB f32
    float*    out  = (float*)d_out;

    // 8192/128 = 64 M-tiles x 4 N-tiles (N-tiles adjacent for L2 reuse of A)
    ft_gemm_kernel<<<256, 256, 0, stream>>>(wfeat, bfeat, ftw, ftb, comb);
    // 4096/128 = 32 M-tiles x 2 N-tiles
    fc1_gemm_kernel<<<64, 256, 0, stream>>>(comb, fc1w, fc1b, xbuf);
    // 4096 rows / 8 waves per block
    fc2_dot_kernel<<<512, 256, 0, stream>>>(xbuf, fc2w, fc2b, out);
}
